// GAPTGN_62173946576946
// MI455X (gfx1250) — compile-verified, matrix operand order checked by reading, not run on hardware
//
#include <hip/hip_runtime.h>
#include <hip/hip_bf16.h>

// ---------------------------------------------------------------------------
// CDNA5 (gfx1250) TGN inference. wave32 blocks, WMMA f32<=f16xf16 throughout.
// Branchless native transcendentals (v_exp/v_rcp/v_tanh/v_cos) to keep the
// VALU side co-executable with the XDL WMMA pipe.
// ---------------------------------------------------------------------------

typedef __attribute__((ext_vector_type(16))) _Float16 v16h;
typedef __attribute__((ext_vector_type(8)))  float    v8f;

union HV { v16h v; _Float16 h[16]; };
union FV { v8f  v; float    f[8];  };

__device__ __forceinline__ v8f wmma_f16(v16h a, v16h b, v8f c) {
  // D = A(16x32 f16) * B(32x16 f16) + C(16x16 f32)
  return __builtin_amdgcn_wmma_f32_16x16x32_f16(
      /*neg_a=*/false, a, /*neg_b=*/false, b,
      /*c_mod=*/(short)0, c, /*reuse_a=*/false, /*reuse_b=*/false);
}

// K index held by half-slot i of a lane, per CDNA5 16-bit A/B VGPR layout:
// lanes 0-15: K {0..7, 16..23}; lanes 16-31: K {8..15, 24..31}
__device__ __forceinline__ int frag_k(int i, int lane) {
  return (i & 7) + ((i >> 3) << 4) + ((lane & 16) ? 8 : 0);
}

// A fragment (16x32) from an f16 LDS tile, row-major, given row stride & k0.
__device__ __forceinline__ v16h load_a_lds_h(const _Float16* t, int stride, int k0, int lane) {
  HV a; const int m = lane & 15;
#pragma unroll
  for (int i = 0; i < 16; ++i) a.h[i] = t[m * stride + k0 + frag_k(i, lane)];
  return a.v;
}
// A fragment from an f32 LDS tile (convert on the fly).
__device__ __forceinline__ v16h load_a_lds_f(const float* t, int stride, int k0, int lane) {
  HV a; const int m = lane & 15;
#pragma unroll
  for (int i = 0; i < 16; ++i) a.h[i] = (_Float16)t[m * stride + k0 + frag_k(i, lane)];
  return a.v;
}
// B fragment (32x16) from global f16 matrix W[k][n], row-major with ncols.
__device__ __forceinline__ v16h load_b_gl(const _Float16* W, int ncols, int k0, int n0, int lane) {
  HV b; const int n = lane & 15;
#pragma unroll
  for (int i = 0; i < 16; ++i)
    b.h[i] = W[(size_t)(k0 + frag_k(i, lane)) * ncols + (n0 + n)];
  return b.v;
}

// ---- branchless native transcendentals (TRANS pipe, co-exec with WMMA) ----
__device__ __forceinline__ float fexp2(float x) { return __builtin_amdgcn_exp2f(x); }
__device__ __forceinline__ float frcp(float x)  { return __builtin_amdgcn_rcpf(x); }
__device__ __forceinline__ float fsig(float x) {
  return frcp(1.f + fexp2(-1.4426950408889634f * x));   // v_exp + v_rcp, no branches
}
__device__ __forceinline__ float ftanh(float x) {
#if __has_builtin(__builtin_amdgcn_tanhf)
  return __builtin_amdgcn_tanhf(x);                      // native v_tanh_f32
#else
  float e = fexp2(2.8853900817779268f * x);              // exp(2x)
  return 1.f - 2.f * frcp(e + 1.f);                      // saturates cleanly at +-1
#endif
}

// ---------------------------------------------------------------------------
// 0) f32 [rows][cols] -> f16 transposed [cols][rows]
// ---------------------------------------------------------------------------
__global__ void k_convT(const float* __restrict__ s, _Float16* __restrict__ d,
                        int rows, int cols) {
  int tid = blockIdx.x * blockDim.x + threadIdx.x;
  if (tid >= rows * cols) return;
  int r = tid / cols, c = tid - r * cols;
  d[(size_t)c * rows + r] = (_Float16)s[tid];
}

// ---------------------------------------------------------------------------
// 1) LastAggregator winner: key = ((t+1)<<32)|j  (max t, tie -> max j: exactly
//    the stable-ascending-argsort rank-max of the reference)
// ---------------------------------------------------------------------------
__global__ void k_winner(const int* __restrict__ srcI, const int* __restrict__ dstI,
                         const int* __restrict__ tI, unsigned long long* __restrict__ win,
                         int E) {
  int j = blockIdx.x * blockDim.x + threadIdx.x;
  if (j >= 2 * E) return;
  int e = (j < E) ? j : (j - E);
  int node = (j < E) ? srcI[e] : dstI[e];
  unsigned long long key =
      ((unsigned long long)(unsigned)(tI[e] + 1) << 32) | (unsigned)j;
  atomicMax(&win[node], key);
}

__global__ void k_active(const unsigned long long* __restrict__ win,
                         int* __restrict__ act, int* __restrict__ cnt, int N) {
  int v = blockIdx.x * blockDim.x + threadIdx.x;
  if (v >= N) return;
  if (win[v] != 0ull) act[atomicAdd(cnt, 1)] = v;
}

// ---------------------------------------------------------------------------
// 2) static projection: src_static[e][d] = b_sp[d] + concat(pe,se) . W_sp[d]
// ---------------------------------------------------------------------------
__global__ void k_static(const int* __restrict__ srcI, const int* __restrict__ xst,
                         const float* __restrict__ pe, const float* __restrict__ se,
                         const float* __restrict__ Wsp, const float* __restrict__ bsp,
                         float* __restrict__ outS, int E) {
  int tid = blockIdx.x * blockDim.x + threadIdx.x;
  if (tid >= E * 128) return;
  int e = tid >> 7, d = tid & 127;
  int s = srcI[e];
  const float* pr = pe + xst[s * 2 + 0] * 16;
  const float* sr = se + xst[s * 2 + 1] * 16;
  const float* w  = Wsp + d * 32;
  float acc = bsp[d];
#pragma unroll
  for (int k = 0; k < 16; ++k) acc += pr[k] * w[k] + sr[k] * w[16 + k];
  outS[tid] = acc;
}

// ---------------------------------------------------------------------------
// 3) Price LSTM: 1 wave == 16 events. Per step: 8 WMMAs (K=32), gates in D
//    layout, h re-staged through LDS as the next step's A fragment.
// ---------------------------------------------------------------------------
__global__ void __launch_bounds__(32) k_lstm(
    const float* __restrict__ price, const float* __restrict__ Wihl,
    const float* __restrict__ bihl, const float* __restrict__ bhhl,
    const _Float16* __restrict__ WhhlT, const _Float16* __restrict__ WppT,
    const float* __restrict__ bpp, float* __restrict__ pemb, int E) {
  const int lane = threadIdx.x;
  const int e0 = blockIdx.x * 16;
  const int n = lane & 15;
  const int mhi = (lane & 16) >> 1;  // +8 row offset for upper lane half

  __shared__ _Float16 hT[16 * 32];
  __shared__ float xs[16];

  v16h Bh[8];
#pragma unroll
  for (int nt = 0; nt < 8; ++nt) Bh[nt] = load_b_gl(WhhlT, 128, 0, nt * 16, lane);
  float wi[8], bs[8];
#pragma unroll
  for (int nt = 0; nt < 8; ++nt) {
    int j = nt * 16 + n;
    wi[nt] = Wihl[j];
    bs[nt] = bihl[j] + bhhl[j];
  }
#pragma unroll
  for (int i = 0; i < 16; ++i) hT[lane * 16 + i] = (_Float16)0.f;
  float cst[16];
#pragma unroll
  for (int i = 0; i < 16; ++i) cst[i] = 0.f;
  __syncthreads();

  for (int step = 0; step < 50; ++step) {
    if (lane < 16) {
      int e = e0 + lane;
      xs[lane] = (e < E) ? price[(size_t)e * 50 + step] : 0.f;
    }
    __syncthreads();
    v16h A = load_a_lds_h(hT, 32, 0, lane);
#pragma unroll
    for (int ct = 0; ct < 2; ++ct) {            // 2 col-halves of LH=32
      FV g[4];
#pragma unroll
      for (int gi = 0; gi < 4; ++gi) {          // i, f, g, o gate tiles
        int nt = gi * 2 + ct;
#pragma unroll
        for (int r = 0; r < 8; ++r) g[gi].f[r] = xs[r + mhi] * wi[nt] + bs[nt];
        g[gi].v = wmma_f16(A, Bh[nt], g[gi].v);
      }
#pragma unroll
      for (int r = 0; r < 8; ++r) {
        float it = fsig(g[0].f[r]);
        float ft = fsig(g[1].f[r]);
        float gt = ftanh(g[2].f[r]);
        float ot = fsig(g[3].f[r]);
        float c = ft * cst[ct * 8 + r] + it * gt;
        cst[ct * 8 + r] = c;
        hT[(r + mhi) * 32 + ct * 16 + n] = (_Float16)(ot * ftanh(c));
      }
    }
    __syncthreads();
  }
  // price_emb = h_n @ W_pp^T + b_pp   (K=32, N=128)
  v16h A = load_a_lds_h(hT, 32, 0, lane);
#pragma unroll
  for (int nt = 0; nt < 8; ++nt) {
    FV acc;
#pragma unroll
    for (int r = 0; r < 8; ++r) acc.f[r] = bpp[nt * 16 + n];
    acc.v = wmma_f16(A, load_b_gl(WppT, 128, 0, nt * 16, lane), acc.v);
#pragma unroll
    for (int r = 0; r < 8; ++r) {
      int e = e0 + r + mhi;
      if (e < E) pemb[(size_t)e * 128 + nt * 16 + n] = acc.f[r];
    }
  }
}

// ---------------------------------------------------------------------------
// 4) GRU update, 32 active nodes per wave (two 16-row A tiles share every B
//    fragment -> half the weight traffic per row). r/z fuse gi+gh
//    accumulation; the n-gate keeps input/hidden parts split.
// ---------------------------------------------------------------------------
__global__ void __launch_bounds__(32) k_gru(
    const int* __restrict__ act, const int* __restrict__ cnt,
    const unsigned long long* __restrict__ win,
    const int* __restrict__ srcI, const int* __restrict__ dstI,
    const float* __restrict__ msg, const float* __restrict__ mem,
    const float* __restrict__ lastu, const float* __restrict__ wt,
    const float* __restrict__ bt,
    const _Float16* __restrict__ WihT, const _Float16* __restrict__ WhhT,
    const float* __restrict__ bih, const float* __restrict__ bhh,
    float* __restrict__ memNew, int E) {
  const int lane = threadIdx.x;
  const int base = blockIdx.x * 32;
  const int count = cnt[0];
  if (base >= count) return;

  __shared__ _Float16 Aagg[32 * 448];   // 28 KB
  __shared__ float Mf[32 * 128];        // 16 KB
  __shared__ int sNode[32], sEvt[32], sOther[32];
  __shared__ float sDt[32];

  {
    int row = base + lane;
    if (row < count) {
      int v = act[row];
      unsigned long long key = win[v];
      unsigned j = (unsigned)(key & 0xffffffffu);
      float tj = (float)((unsigned)(key >> 32) - 1u);
      int e, other;
      if ((int)j < E) { e = (int)j;     other = dstI[e]; }
      else            { e = (int)j - E; other = srcI[e]; }
      sNode[lane] = v; sEvt[lane] = e; sOther[lane] = other;
      sDt[lane] = tj - lastu[v];
    } else {
      sNode[lane] = -1;
    }
  }
  // warm L2 with the weight block this wave is about to stream repeatedly
  __builtin_prefetch(WihT + (size_t)lane * 64, 0, 1);
  __syncthreads();
  for (int idx = lane; idx < 32 * 448; idx += 32) {
    int m = idx / 448, c = idx - m * 448;
    int v = sNode[m];
    float val = 0.f;
    if (v >= 0) {
      if (c < 128)      { val = mem[(size_t)v * 128 + c]; Mf[m * 128 + c] = val; }
      else if (c < 256)   val = mem[(size_t)sOther[m] * 128 + (c - 128)];
      else if (c < 320)   val = msg[(size_t)sEvt[m] * 64 + (c - 256)];
      else              { int q = c - 320; val = __cosf(sDt[m] * wt[q] + bt[q]); }
    } else if (c < 128) Mf[m * 128 + c] = 0.f;
    Aagg[idx] = (_Float16)val;
  }
  __syncthreads();

  const int n = lane & 15;
  const int mhi = (lane & 16) >> 1;
#pragma unroll 1
  for (int cb = 0; cb < 8; ++cb) {             // 8 x 16 = 128 output dims
    const int j0 = cb * 16 + n;
    FV ar0, az0, ai0, ah0, ar1, az1, ai1, ah1;
    {
      const float br = bih[j0] + bhh[j0];
      const float bz = bih[128 + j0] + bhh[128 + j0];
      const float bi = bih[256 + j0];
      const float bh = bhh[256 + j0];
#pragma unroll
      for (int r = 0; r < 8; ++r) {
        ar0.f[r] = br; az0.f[r] = bz; ai0.f[r] = bi; ah0.f[r] = bh;
        ar1.f[r] = br; az1.f[r] = bz; ai1.f[r] = bi; ah1.f[r] = bh;
      }
    }
    for (int k = 0; k < 14; ++k) {             // K=448 over agg message
      v16h A0 = load_a_lds_h(Aagg,            448, k * 32, lane);
      v16h A1 = load_a_lds_h(Aagg + 16 * 448, 448, k * 32, lane);
      v16h Br = load_b_gl(WihT, 384, k * 32,       cb * 16, lane);
      v16h Bz = load_b_gl(WihT, 384, k * 32, 128 + cb * 16, lane);
      v16h Bi = load_b_gl(WihT, 384, k * 32, 256 + cb * 16, lane);
      ar0.v = wmma_f16(A0, Br, ar0.v);  ar1.v = wmma_f16(A1, Br, ar1.v);
      az0.v = wmma_f16(A0, Bz, az0.v);  az1.v = wmma_f16(A1, Bz, az1.v);
      ai0.v = wmma_f16(A0, Bi, ai0.v);  ai1.v = wmma_f16(A1, Bi, ai1.v);
    }
    for (int k = 0; k < 4; ++k) {              // K=128 over old memory
      v16h A0 = load_a_lds_f(Mf,            128, k * 32, lane);
      v16h A1 = load_a_lds_f(Mf + 16 * 128, 128, k * 32, lane);
      v16h Br = load_b_gl(WhhT, 384, k * 32,       cb * 16, lane);
      v16h Bz = load_b_gl(WhhT, 384, k * 32, 128 + cb * 16, lane);
      v16h Bh = load_b_gl(WhhT, 384, k * 32, 256 + cb * 16, lane);
      ar0.v = wmma_f16(A0, Br, ar0.v);  ar1.v = wmma_f16(A1, Br, ar1.v);
      az0.v = wmma_f16(A0, Bz, az0.v);  az1.v = wmma_f16(A1, Bz, az1.v);
      ah0.v = wmma_f16(A0, Bh, ah0.v);  ah1.v = wmma_f16(A1, Bh, ah1.v);
    }
#pragma unroll
    for (int tile = 0; tile < 2; ++tile) {
      FV& ar = tile ? ar1 : ar0;
      FV& az = tile ? az1 : az0;
      FV& ai = tile ? ai1 : ai0;
      FV& ah = tile ? ah1 : ah0;
#pragma unroll
      for (int r = 0; r < 8; ++r) {
        int m = tile * 16 + r + mhi;
        if (base + m < count) {
          float rr = fsig(ar.f[r]);
          float zz = fsig(az.f[r]);
          float nn = ftanh(ai.f[r] + rr * ah.f[r]);
          float mo = Mf[m * 128 + j0];
          memNew[(size_t)sNode[m] * 128 + j0] = (1.f - zz) * nn + zz * mo;
        }
      }
    }
  }
}

// ---------------------------------------------------------------------------
// 5) Fuse + predictor MLP: combined[16x384] -> relu(@W1^T) -> @W2 + b2
// ---------------------------------------------------------------------------
__global__ void __launch_bounds__(32) k_mlp(
    const int* __restrict__ srcI, const int* __restrict__ dstI,
    const float* __restrict__ memNew, const float* __restrict__ sstat,
    const float* __restrict__ pemb,
    const _Float16* __restrict__ W1T, const float* __restrict__ b1,
    const float* __restrict__ W2, const float* __restrict__ b2,
    float* __restrict__ out, int E) {
  const int lane = threadIdx.x;
  const int e0 = blockIdx.x * 16;
  __shared__ _Float16 C[16 * 384];
  __shared__ float H1[16 * 64];
  for (int idx = lane; idx < 16 * 384; idx += 32) {
    int m = idx / 384, c = idx - m * 384;
    int e = e0 + m;
    float val = 0.f;
    if (e < E) {
      if (c < 128)
        val = memNew[(size_t)srcI[e] * 128 + c] + sstat[(size_t)e * 128 + c];
      else if (c < 256) {
        int cc = c - 128;
        val = memNew[(size_t)dstI[e] * 128 + cc] + pemb[(size_t)e * 128 + cc];
      } else
        val = pemb[(size_t)e * 128 + (c - 256)];
    }
    C[idx] = (_Float16)val;
  }
  __syncthreads();
  const int n = lane & 15;
  const int mhi = (lane & 16) >> 1;
#pragma unroll
  for (int nt = 0; nt < 4; ++nt) {             // N=64
    FV acc;
#pragma unroll
    for (int r = 0; r < 8; ++r) acc.f[r] = b1[nt * 16 + n];
    for (int k = 0; k < 12; ++k) {             // K=384
      v16h A = load_a_lds_h(C, 384, k * 32, lane);
      acc.v = wmma_f16(A, load_b_gl(W1T, 64, k * 32, nt * 16, lane), acc.v);
    }
#pragma unroll
    for (int r = 0; r < 8; ++r)
      H1[(r + mhi) * 64 + nt * 16 + n] = fmaxf(acc.f[r], 0.f);
  }
  __syncthreads();
  if (lane < 16) {
    int e = e0 + lane;
    if (e < E) {
      float s = b2[0];
#pragma unroll
      for (int q = 0; q < 64; ++q) s += H1[lane * 64 + q] * W2[q];
      out[e] = s;
    }
  }
}

// ---------------------------------------------------------------------------
extern "C" void kernel_launch(void* const* d_in, const int* in_sizes, int n_in,
                              void* d_out, int out_size, void* d_ws, size_t ws_size,
                              hipStream_t stream) {
  (void)n_in; (void)out_size; (void)ws_size;
  const int E = in_sizes[0];
  const int N = in_sizes[8];

  const int*   srcI  = (const int*)d_in[0];
  const int*   dstI  = (const int*)d_in[1];
  const int*   tI    = (const int*)d_in[2];
  const float* msg   = (const float*)d_in[3];
  const float* price = (const float*)d_in[4];
  const int*   xst   = (const int*)d_in[6];
  const float* mem   = (const float*)d_in[7];
  const float* lastu = (const float*)d_in[8];
  const float* wt    = (const float*)d_in[9];
  const float* bt    = (const float*)d_in[10];
  const float* Wih   = (const float*)d_in[11];
  const float* bih   = (const float*)d_in[12];
  const float* Whh   = (const float*)d_in[13];
  const float* bhh   = (const float*)d_in[14];
  const float* pe    = (const float*)d_in[15];
  const float* se    = (const float*)d_in[16];
  const float* Wsp   = (const float*)d_in[17];
  const float* bsp   = (const float*)d_in[18];
  const float* Wihl  = (const float*)d_in[19];
  const float* Whhl  = (const float*)d_in[20];
  const float* bihl  = (const float*)d_in[21];
  const float* bhhl  = (const float*)d_in[22];
  const float* Wpp   = (const float*)d_in[23];
  const float* bpp   = (const float*)d_in[24];
  const float* W1    = (const float*)d_in[25];
  const float* b1    = (const float*)d_in[26];
  const float* W2    = (const float*)d_in[27];
  const float* b2    = (const float*)d_in[28];
  float* out = (float*)d_out;

  // ---- workspace layout ----
  char* p = (char*)d_ws;
  auto take = [&](size_t b) { void* r = (void*)p; p += (b + 255) & ~(size_t)255; return r; };
  unsigned long long* win = (unsigned long long*)take((size_t)N * 8);
  int*   cnt    = (int*)take(256);
  int*   act    = (int*)take((size_t)N * 4);
  float* memNew = (float*)take((size_t)N * 128 * 4);
  float* pemb   = (float*)take((size_t)E * 128 * 4);
  float* sstat  = (float*)take((size_t)E * 128 * 4);
  _Float16* WihT  = (_Float16*)take((size_t)448 * 384 * 2);
  _Float16* WhhT  = (_Float16*)take((size_t)128 * 384 * 2);
  _Float16* W1T   = (_Float16*)take((size_t)384 * 64 * 2);
  _Float16* WhhlT = (_Float16*)take((size_t)32 * 128 * 2);
  _Float16* WppT  = (_Float16*)take((size_t)32 * 128 * 2);

  hipMemsetAsync(win, 0, (size_t)N * 8, stream);
  hipMemsetAsync(cnt, 0, sizeof(int), stream);

  // weight prep (cheap; recomputed every call for determinism under capture)
  k_convT<<<(384 * 448 + 255) / 256, 256, 0, stream>>>(Wih,  WihT,  384, 448);
  k_convT<<<(384 * 128 + 255) / 256, 256, 0, stream>>>(Whh,  WhhT,  384, 128);
  k_convT<<<(64 * 384 + 255) / 256, 256, 0, stream>>>(W1,   W1T,   64,  384);
  k_convT<<<(128 * 32 + 255) / 256, 256, 0, stream>>>(Whhl, WhhlT, 128, 32);
  k_convT<<<(128 * 32 + 255) / 256, 256, 0, stream>>>(Wpp,  WppT,  128, 32);

  k_winner<<<(2 * E + 255) / 256, 256, 0, stream>>>(srcI, dstI, tI, win, E);
  k_active<<<(N + 255) / 256, 256, 0, stream>>>(win, act, cnt, N);
  k_static<<<(E * 128 + 255) / 256, 256, 0, stream>>>(srcI, xst, pe, se, Wsp, bsp, sstat, E);
  k_lstm<<<(E + 15) / 16, 32, 0, stream>>>(price, Wihl, bihl, bhhl, WhhlT, WppT, bpp, pemb, E);
  k_gru<<<(N + 31) / 32, 32, 0, stream>>>(act, cnt, win, srcI, dstI, msg, mem, lastu,
                                          wt, bt, WihT, WhhT, bih, bhh, memNew, E);
  k_mlp<<<(E + 15) / 16, 32, 0, stream>>>(srcI, dstI, memNew, sstat, pemb,
                                          W1T, b1, W2, b2, out, E);
}
